// Pairwise_29274497089740
// MI455X (gfx1250) — compile-verified, matrix operand order checked by reading, not run on hardware
//
#include <hip/hip_runtime.h>

// Problem constants: B=8, C=512, H=W=96, 5x5 context,
// conv1: 25 -> 256, conv2: 256 -> 128, output = concat(ss, ss1, ss2) = 409 ch.
#define BATCH   8
#define CFEAT   512
#define HH      96
#define WW      96
#define HW      (HH * WW)          // 9216
#define NCH_OUT 409
#define BSTRIDE ((size_t)NCH_OUT * HW)
#define EPSN    1e-12f
#define XT      (WW / 16)          // 6 pixel tiles per row

typedef __attribute__((ext_vector_type(16))) _Float16 v16h;
typedef __attribute__((ext_vector_type(8)))  _Float16 v8h;
typedef __attribute__((ext_vector_type(8)))  float    v8f;

// ---------------------------------------------------------------------------
// Kernel 1: per-pixel inverse L2 norm over 512 channels.
// ---------------------------------------------------------------------------
__global__ void invnorm_kernel(const float* __restrict__ feat,
                               float* __restrict__ invn) {
    int p = blockIdx.x * blockDim.x + threadIdx.x;
    if (p >= BATCH * HW) return;
    int b = p / HW;
    int rem = p - b * HW;
    const float* f = feat + (size_t)b * CFEAT * HW + rem;
    float s = 0.f;
#pragma unroll 8
    for (int c = 0; c < CFEAT; ++c) {
        float v = f[(size_t)c * HW];
        s = fmaf(v, v, s);
    }
    invn[p] = 1.f / fmaxf(sqrtf(s), EPSN);
}

// ---------------------------------------------------------------------------
// Kernel 2: 25-plane self-similarity on RAW features; normalization applied
// as the bilinear scalar inv_p * inv_q in the epilogue (never materializes fn).
// Writes channels [0,25) of the concatenated output.
// ---------------------------------------------------------------------------
#define CHUNK 4
__global__ void selfsim_kernel(const float* __restrict__ feat,
                               const float* __restrict__ invn,
                               float* __restrict__ out) {
    __shared__ float patch[CHUNK][20][20];
    const int tx = threadIdx.x, ty = threadIdx.y;
    const int x0 = blockIdx.x * 16, y0 = blockIdx.y * 16, b = blockIdx.z;
    const int x = x0 + tx, y = y0 + ty;
    const int tid = ty * 16 + tx;

    float acc[25];
#pragma unroll
    for (int i = 0; i < 25; ++i) acc[i] = 0.f;

    const float* fb = feat + (size_t)b * CFEAT * HW;

    for (int c0 = 0; c0 < CFEAT; c0 += CHUNK) {
        for (int idx = tid; idx < CHUNK * 400; idx += 256) {
            int c  = idx / 400;
            int r  = idx - c * 400;
            int yy = r / 20;
            int xx = r - yy * 20;
            int gy = y0 + yy - 2, gx = x0 + xx - 2;
            float v = 0.f;
            if (gy >= 0 && gy < HH && gx >= 0 && gx < WW)
                v = fb[(size_t)(c0 + c) * HW + gy * WW + gx];
            patch[c][yy][xx] = v;
        }
        __syncthreads();
#pragma unroll
        for (int c = 0; c < CHUNK; ++c) {
            float cen = patch[c][ty + 2][tx + 2];
#pragma unroll
            for (int cc = 0; cc < 5; ++cc)
#pragma unroll
                for (int r = 0; r < 5; ++r)
                    acc[cc * 5 + r] = fmaf(patch[c][ty + r][tx + cc], cen,
                                           acc[cc * 5 + r]);
        }
        __syncthreads();
    }

    const float ic = invn[b * HW + y * WW + x];
    float* ob = out + (size_t)b * BSTRIDE + (size_t)y * WW + x;
#pragma unroll
    for (int cc = 0; cc < 5; ++cc) {
#pragma unroll
        for (int r = 0; r < 5; ++r) {
            int ny = y + r - 2, nx = x + cc - 2;
            float iq = (ny >= 0 && ny < HH && nx >= 0 && nx < WW)
                           ? invn[b * HW + ny * WW + nx] : 0.f;
            ob[(size_t)(cc * 5 + r) * HW] = acc[cc * 5 + r] * ic * iq;
        }
    }
}

// ---------------------------------------------------------------------------
// Weight prepack: w [COUT][CIN][3][3] fp32 -> f16 in exact A-fragment register
// order for v_wmma_f32_16x16x32_f16, with K = (ky*3+kx)*CINP + cin
// (spatial-major, channel-minor, channels zero-padded CIN->CINP).
// apk layout: [cout_tile][kc][lane][16 halves] (32 B per lane, contiguous).
// ---------------------------------------------------------------------------
template <int CIN, int CINP, int COUT>
__global__ void pack_weights_kernel(const float* __restrict__ w,
                                    _Float16* __restrict__ apk) {
    constexpr int CB = CINP / 32;
    constexpr int KC = 9 * CB;
    constexpr int NT = COUT / 16;
    int idx = blockIdx.x * blockDim.x + threadIdx.x;
    if (idx >= NT * KC * 32) return;
    int lane = idx & 31;
    int kc   = (idx >> 5) % KC;
    int tile = idx / (32 * KC);
    int lm = lane & 15, half = lane >> 4;
    int r = kc / CB;                  // ky*3+kx
    int cbase = (kc % CB) * 32;
    int m = tile * 16 + lm;

    v16h vals;
#pragma unroll
    for (int i = 0; i < 16; ++i) {
        int cin = cbase + i + ((i >= 8) ? 8 : 0) + 8 * half;
        float v = (cin < CIN) ? w[((size_t)m * CIN + cin) * 9 + r] : 0.f;
        vals[i] = (_Float16)v;
    }
    *(v16h*)(apk + (size_t)idx * 16) = vals;
}

// ---------------------------------------------------------------------------
// 3x3 SAME conv + bias + ReLU, implicit GEMM on v_wmma_f32_16x16x32_f16.
//   Wave tile: 16 Cout x 16 x-pixels.  Block: 8 waves sharing one LDS halo
//   tile [3][18][CINP] (f16, channel stride padded +8 halves for LDS banking).
//   Inner step (fully unrolled, 9*CINP/32 WMMAs):
//     B frag = 2x ds_load_b128 (contiguous channel runs) + shuffle
//     A frag = one 32B coalesced global load of prepacked weights
// ---------------------------------------------------------------------------
template <int CIN, int CINP, int COUT>
__global__ void conv3x3_wmma_kernel(const float* __restrict__ in_base,
                                    const _Float16* __restrict__ apk,
                                    const float* __restrict__ bias,
                                    float* __restrict__ out_base) {
    constexpr int CB = CINP / 32;
    constexpr int KC = 9 * CB;
    constexpr int S  = CINP + 8;          // padded channel stride (halves)
    __shared__ _Float16 tile[3 * 18 * S];

    const int ln = threadIdx.x & 31;
    const int wv = threadIdx.x >> 5;
    const int lm = ln & 15;
    const int half = ln >> 4;
    const int ctile = blockIdx.y * 8 + wv;        // cout tile index

    int t  = blockIdx.x;
    int xt = t % XT;
    int yz = t / XT;
    int oy = yz % HH;
    int b  = yz / HH;
    const int x0 = xt * 16;

    // ---- stage f16 halo tile: rows oy-1..oy+1, cols x0-1..x0+16, CINP ch ----
    const float* inb = in_base + (size_t)b * BSTRIDE;
    for (int c = wv; c < CINP; c += 8) {
#pragma unroll
        for (int pos = ln; pos < 54; pos += 32) {
            int row = pos / 18, col = pos - row * 18;
            int gy = oy + row - 1, gx = x0 + col - 1;
            float v = 0.f;
            if ((CIN == CINP || c < CIN) &&
                gy >= 0 && gy < HH && gx >= 0 && gx < WW)
                v = inb[(size_t)c * HW + gy * WW + gx];
            tile[(row * 18 + col) * S + c] = (_Float16)v;
        }
    }
    __syncthreads();

    // ---- K loop: r = ky*3+kx outer, 32-channel blocks inner ----
    const _Float16* ap = apk + (size_t)ctile * KC * 512 + (size_t)ln * 16;
    v8f acc = {};
    int kc = 0;
#pragma unroll
    for (int ky = 0; ky < 3; ++ky) {
#pragma unroll
        for (int kx = 0; kx < 3; ++kx) {
            const _Float16* bb = &tile[(ky * 18 + lm + kx) * S + 8 * half];
#pragma unroll
            for (int cb = 0; cb < CINP; cb += 32) {
                v8h blo = *(const v8h*)(bb + cb);        // ds_load_b128
                v8h bhi = *(const v8h*)(bb + cb + 16);   // ds_load_b128
                v16h bf = __builtin_shufflevector(
                    blo, bhi, 0, 1, 2, 3, 4, 5, 6, 7,
                    8, 9, 10, 11, 12, 13, 14, 15);
                v16h af = *(const v16h*)(ap + (size_t)kc * 512);  // 32B / lane
                acc = __builtin_amdgcn_wmma_f32_16x16x32_f16(
                    false, af, false, bf, (short)0, acc, false, false);
                ++kc;
            }
        }
    }

    // ---- epilogue: bias + ReLU, coalesced row stores ----
    float* ob = out_base + (size_t)b * BSTRIDE + (size_t)oy * WW + (x0 + lm);
#pragma unroll
    for (int r = 0; r < 8; ++r) {
        int m = ctile * 16 + r + 8 * half;
        float v = acc[r] + bias[m];
        ob[(size_t)m * HW] = fmaxf(v, 0.f);
    }
}

// ---------------------------------------------------------------------------
extern "C" void kernel_launch(void* const* d_in, const int* in_sizes, int n_in,
                              void* d_out, int out_size, void* d_ws, size_t ws_size,
                              hipStream_t stream) {
    (void)in_sizes; (void)n_in; (void)out_size; (void)ws_size;
    const float* feat = (const float*)d_in[0];
    const float* w0   = (const float*)d_in[1];
    const float* b0   = (const float*)d_in[2];
    const float* w1   = (const float*)d_in[3];
    const float* b1   = (const float*)d_in[4];
    float* out = (float*)d_out;

    // Workspace layout (all 64B aligned):
    //   invn : BATCH*HW floats             =   294,912 B
    //   apk1 : 16 tiles * 9 kc * 512 f16   =   147,456 B
    //   apk2 : 8 tiles * 72 kc * 512 f16   =   589,824 B
    char* ws = (char*)d_ws;
    float*     invn = (float*)ws;
    _Float16*  apk1 = (_Float16*)(ws + 2949120);
    _Float16*  apk2 = (_Float16*)(ws + 2949120 + 147456);

    // 1) per-pixel inverse norms
    invnorm_kernel<<<(BATCH * HW + 255) / 256, 256, 0, stream>>>(feat, invn);

    // 2) prepack conv weights into WMMA A-fragment order
    pack_weights_kernel<25, 32, 256><<<(16 * 9 * 32 + 255) / 256, 256, 0, stream>>>(w0, apk1);
    pack_weights_kernel<256, 256, 128><<<(8 * 72 * 32 + 255) / 256, 256, 0, stream>>>(w1, apk2);

    // 3) cost volume -> output channels [0,25)
    selfsim_kernel<<<dim3(XT, HH / 16, BATCH), dim3(16, 16), 0, stream>>>(
        feat, invn, out);

    // 4) conv1 (25 -> 256): reads ss at ch 0, writes ss1 at ch 25
    conv3x3_wmma_kernel<25, 32, 256><<<dim3(BATCH * HH * XT, 2), 256, 0, stream>>>(
        out, apk1, b0, out + (size_t)25 * HW);

    // 5) conv2 (256 -> 128): reads ss1 at ch 25, writes ss2 at ch 281
    conv3x3_wmma_kernel<256, 256, 128><<<dim3(BATCH * HH * XT, 1), 256, 0, stream>>>(
        out + (size_t)25 * HW, apk2, b1, out + (size_t)281 * HW);
}
